// PairUpdate_50044958933075
// MI455X (gfx1250) — compile-verified, matrix operand order checked by reading, not run on hardware
//
// CDNA5 (gfx1250, MI455X) implementation of the PairUpdate block.
// All dense contractions are routed through v_wmma_f32_16x16x32_f16.
// wave32; fragments use the exact ISA VGPR layouts from CDNA5 ISA.
#include <hip/hip_runtime.h>
#include <math.h>

typedef _Float16 h16;
typedef __attribute__((ext_vector_type(16))) _Float16 v16h;
typedef __attribute__((ext_vector_type(8)))  _Float16 v8h;
typedef __attribute__((ext_vector_type(8)))  float    v8f;

#define DI 384
#define DJ 128
#define R1 (DI*DJ)   // 49152 z rows
#define R2 (DI*DI)   // 147456 pocket rows
#define R3 (DJ*DJ)   // 16384 mol rows

__device__ __forceinline__ v8f wmma16(v16h a, v16h b, v8f c) {
  return __builtin_amdgcn_wmma_f32_16x16x32_f16(false, a, false, b, (short)0, c,
                                                false, false);
}

__device__ __forceinline__ float sigm(float x) { return 1.f / (1.f + __expf(-x)); }

// Load a 16x32 f16 fragment (A-matrix layout; also valid for B when the
// "B" operand is stored as [N,K] row-major, i.e. W of x@W^T).
// lane<16: row=lane, K chunks [0..7],[16..23]; lane>=16: row=lane-16, [8..15],[24..31]
__device__ __forceinline__ v16h frag_ld(const h16* __restrict__ base, int ldr) {
  int lane = threadIdx.x & 31;
  const h16* rp = base + (size_t)(lane & 15) * (size_t)ldr + ((lane >> 4) << 3);
  v8h lo = *(const v8h*)(rp);
  v8h hi = *(const v8h*)(rp + 16);
  v16h r;
#pragma unroll
  for (int t = 0; t < 8; ++t) { r[t] = lo[t]; r[t + 8] = hi[t]; }
  return r;
}

// ---------------- elementwise / LN helpers ----------------

__global__ void k_f32_to_f16(const float* __restrict__ in, h16* __restrict__ out, int n) {
  int t = blockIdx.x * blockDim.x + threadIdx.x;
  if (t < n) out[t] = (h16)in[t];
}

// LayerNorm over last dim 128 (biased variance), one wave per row, f32 in -> f16 out.
__global__ void k_ln(const float* __restrict__ x, const float* __restrict__ g,
                     const float* __restrict__ b, h16* __restrict__ out, int R) {
  int row = blockIdx.x * (blockDim.x >> 5) + (threadIdx.x >> 5);
  if (row >= R) return;
  int lane = threadIdx.x & 31;
  const float* rp = x + (size_t)row * 128;
  float v[4], s = 0.f, s2 = 0.f;
#pragma unroll
  for (int t = 0; t < 4; ++t) { float q = rp[lane + 32 * t]; v[t] = q; s += q; s2 += q * q; }
#pragma unroll
  for (int m = 16; m; m >>= 1) { s += __shfl_xor(s, m, 32); s2 += __shfl_xor(s2, m, 32); }
  float mean = s * (1.f / 128.f);
  float var  = s2 * (1.f / 128.f) - mean * mean;
  float rstd = rsqrtf(var + 1e-5f);
  h16* op = out + (size_t)row * 128;
#pragma unroll
  for (int t = 0; t < 4; ++t) {
    int c = lane + 32 * t;
    op[c] = (h16)((v[t] - mean) * rstd * g[c] + b[c]);
  }
}

// zc = g * att  (f16 <- f16 * f32)
__global__ void k_gmul(const h16* __restrict__ g, const float* __restrict__ a,
                       h16* __restrict__ out, int n) {
  int t = blockIdx.x * blockDim.x + threadIdx.x;
  if (t < n) out[t] = (h16)((float)g[t] * a[t]);
}

// ---------------- layout permutes ----------------

// out[c*R + r] = in[r*128 + c]  (channel-major transpose)
__global__ void k_tr_cmajor(const h16* __restrict__ in, h16* __restrict__ out, int R) {
  size_t t = (size_t)blockIdx.x * blockDim.x + threadIdx.x;
  if (t >= (size_t)R * 128) return;
  int    c = (int)(t / (size_t)R);
  size_t r = t % (size_t)R;
  out[t] = in[r * 128 + c];
}

// AB1t[c][j][k] = ab1[(k*128+j)*128 + c]   (c:128, j:128, k:384)
__global__ void k_perm_ab1(const h16* __restrict__ in, h16* __restrict__ out) {
  size_t t = (size_t)blockIdx.x * blockDim.x + threadIdx.x;
  if (t >= (size_t)R1 * 128) return;
  int k = (int)(t % 384); size_t u = t / 384;
  int j = (int)(u % 128); int c = (int)(u / 128);
  out[t] = in[((size_t)k * 128 + j) * 128 + c];
}

// vT1[((i*4+h)*32+c)*128 + j] = v[(i*128+j)*128 + h*32 + c]
__global__ void k_perm_vT1(const h16* __restrict__ in, h16* __restrict__ out) {
  size_t t = (size_t)blockIdx.x * blockDim.x + threadIdx.x;
  if (t >= (size_t)R1 * 128) return;
  int j = (int)(t & 127), c = (int)((t >> 7) & 31), h = (int)((t >> 12) & 3);
  int i = (int)(t >> 14);
  out[t] = in[((size_t)i * 128 + j) * 128 + h * 32 + c];
}

// vT2[((j*4+h)*32+c)*384 + i] = v[(i*128+j)*128 + h*32 + c]
__global__ void k_perm_vT2(const h16* __restrict__ in, h16* __restrict__ out) {
  size_t t = (size_t)blockIdx.x * blockDim.x + threadIdx.x;
  if (t >= (size_t)R1 * 128) return;
  int i = (int)(t % 384); size_t u = t / 384;
  int c = (int)(u & 31), h = (int)((u >> 5) & 3), j = (int)(u >> 7);
  out[t] = in[((size_t)i * 128 + j) * 128 + h * 32 + c];
}

// ---------------- WMMA GEMM workhorse ----------------
// C[M,N] = epi(A[M,K] @ W[N,K]^T + bias)
// MODE 0: f16 out           1: f16 sigmoid      2: f16 relu
//      3: f16 *scale (q)    4: f32 out (final)
//      5: f32 resid + v*mask[m]     6: f32 resid + gate*v*mask[m]
template <int MODE>
__global__ void k_gemm(const h16* __restrict__ A, const h16* __restrict__ W,
                       const float* __restrict__ bias, void* __restrict__ out,
                       int M, int N, int K,
                       const float* __restrict__ resid, const h16* __restrict__ gate,
                       const float* __restrict__ maskrow, float scale) {
  int wave = threadIdx.x >> 5, lane = threadIdx.x & 31;
  int m0 = (blockIdx.x * 4 + wave) * 16;
  int n0 = blockIdx.y * 64;
  if (m0 >= M) return;                       // wave-uniform
  v8f acc[4] = {};
  for (int k0 = 0; k0 < K; k0 += 32) {
    v16h a = frag_ld(A + (size_t)m0 * K + k0, K);
#pragma unroll
    for (int t = 0; t < 4; ++t) {
      v16h w = frag_ld(W + (size_t)(n0 + 16 * t) * K + k0, K);
      acc[t] = wmma16(a, w, acc[t]);
    }
  }
  int n_l = lane & 15, m_l = (lane >> 4) << 3;
#pragma unroll
  for (int t = 0; t < 4; ++t) {
    int n = n0 + 16 * t + n_l;
    float bv = bias ? bias[n] : 0.f;
#pragma unroll
    for (int r = 0; r < 8; ++r) {
      int m = m0 + m_l + r;
      size_t o = (size_t)m * N + n;
      float v = acc[t][r] + bv;
      if (MODE == 0) ((h16*)out)[o] = (h16)v;
      else if (MODE == 1) ((h16*)out)[o] = (h16)sigm(v);
      else if (MODE == 2) ((h16*)out)[o] = (h16)fmaxf(v, 0.f);
      else if (MODE == 3) ((h16*)out)[o] = (h16)(v * scale);
      else if (MODE == 4) ((float*)out)[o] = v;
      else if (MODE == 5) ((float*)out)[o] = resid[o] + v * maskrow[m];
      else if (MODE == 6) ((float*)out)[o] = resid[o] + (float)gate[o] * v * maskrow[m];
    }
  }
}

// out = sigmoid(A@Wg^T+bg) * (A@Wl^T+bl) * mask[m]   (N=K=128), f16 out
__global__ void k_dualgemm(const h16* __restrict__ A,
                           const h16* __restrict__ Wg, const float* __restrict__ bg,
                           const h16* __restrict__ Wl, const float* __restrict__ bl,
                           h16* __restrict__ out, const float* __restrict__ maskrow) {
  int wave = threadIdx.x >> 5, lane = threadIdx.x & 31;
  int m0 = (blockIdx.x * 4 + wave) * 16;
  int n0 = blockIdx.y * 32;
  v8f ag[2] = {}, al[2] = {};
  for (int k0 = 0; k0 < 128; k0 += 32) {
    v16h a = frag_ld(A + (size_t)m0 * 128 + k0, 128);
#pragma unroll
    for (int t = 0; t < 2; ++t) {
      v16h wg = frag_ld(Wg + (size_t)(n0 + 16 * t) * 128 + k0, 128);
      ag[t] = wmma16(a, wg, ag[t]);
      v16h wl = frag_ld(Wl + (size_t)(n0 + 16 * t) * 128 + k0, 128);
      al[t] = wmma16(a, wl, al[t]);
    }
  }
  int n_l = lane & 15, m_l = (lane >> 4) << 3;
#pragma unroll
  for (int t = 0; t < 2; ++t) {
    int n = n0 + 16 * t + n_l;
#pragma unroll
    for (int r = 0; r < 8; ++r) {
      int m = m0 + m_l + r;
      float g = sigm(ag[t][r] + bg[n]);
      float l = al[t][r] + bl[n];
      float mk = maskrow ? maskrow[m] : 1.f;
      out[(size_t)m * 128 + n] = (h16)(g * l * mk);
    }
  }
}

// Per-channel triangle GEMM: out[(m*128+n)*128+c] (beta?+=:=) A_c[384,K] @ W_c[128,K]^T
__global__ void k_tri(const h16* __restrict__ Ab, const h16* __restrict__ Wb,
                      float* __restrict__ out, int K, int beta) {
  int c = blockIdx.z;
  const h16* A = Ab + (size_t)c * 384 * K;
  const h16* W = Wb + (size_t)c * 128 * K;
  int wave = threadIdx.x >> 5, lane = threadIdx.x & 31;
  int m0 = (blockIdx.x * 4 + wave) * 16;    // 6 blocks -> 384 rows
  int n0 = blockIdx.y * 64;                 // 2 blocks -> 128 cols
  v8f acc[4] = {};
  for (int k0 = 0; k0 < K; k0 += 32) {
    v16h a = frag_ld(A + (size_t)m0 * K + k0, K);
#pragma unroll
    for (int t = 0; t < 4; ++t) {
      v16h w = frag_ld(W + (size_t)(n0 + 16 * t) * K + k0, K);
      acc[t] = wmma16(a, w, acc[t]);
    }
  }
  int n_l = lane & 15, m_l = (lane >> 4) << 3;
#pragma unroll
  for (int t = 0; t < 4; ++t)
#pragma unroll
    for (int r = 0; r < 8; ++r) {
      size_t o = ((size_t)(m0 + m_l + r) * 128 + (n0 + 16 * t + n_l)) * 128 + c;
      if (beta) out[o] += acc[t][r]; else out[o] = acc[t][r];
    }
}

// ---------------- axis attention ----------------
// pass 1: per (i,h) attend over J=128 keys; pass 2: per (j,h) attend over I=384 keys.
// block = 64 threads (2 waves); each wave streams 16-query tiles.
// Vt is pre-transposed [outer][h][c(32)][key(NK)] so P@V is the same [M,K]x[N,K] form.
__global__ void k_attn(const h16* __restrict__ Q, const h16* __restrict__ Kb,
                       const h16* __restrict__ Vt, const float* __restrict__ mask,
                       float* __restrict__ out, int pass, int beta) {
  int outer = blockIdx.x, h = blockIdx.y;
  int wave = threadIdx.x >> 5, lane = threadIdx.x & 31;
  int NK = (pass == 1) ? 128 : 384;
  int NQ = NK;
  size_t qk_off; int rs; size_t vt_off; const float* bias; int bstride;
  if (pass == 1) {
    qk_off = (size_t)outer * 128 * 128 + (size_t)h * 32; rs = 128;
    vt_off = ((size_t)outer * 4 + h) * 32 * 128;
    bias = mask + (size_t)outer * 128; bstride = 1;
  } else {
    qk_off = (size_t)outer * 128 + (size_t)h * 32; rs = 16384;
    vt_off = ((size_t)outer * 4 + h) * 32 * 384;
    bias = mask + outer; bstride = 128;
  }
  size_t out_off = qk_off; int out_rs = rs;   // same addressing as Q rows
  extern __shared__ float smem[];
  float* sc = smem + (size_t)wave * 16 * NK;
  int n_l = lane & 15, m_l = (lane >> 4) << 3;
  int nqt = NQ / 32;                          // tiles per wave (2 waves/block)
  for (int qt = 0; qt < nqt; ++qt) {
    int q0 = (qt * 2 + wave) * 16;
    // ---- scores = Q(16x32) @ K^T + bias ----
    v16h aq = frag_ld(Q + qk_off + (size_t)q0 * rs, rs);
    for (int kt = 0; kt < NK / 16; ++kt) {
      v16h bk = frag_ld(Kb + qk_off + (size_t)(kt * 16) * rs, rs);
      v8f s = {};
      s = wmma16(aq, bk, s);
      int kk = kt * 16 + n_l;
      float bv = 1e9f * (bias[(size_t)kk * bstride] - 1.f);
#pragma unroll
      for (int r = 0; r < 8; ++r) sc[(m_l + r) * NK + kk] = s[r] + bv;
    }
    __syncthreads();
    // ---- softmax (lanes 0..15 own one row each) ----
    if (lane < 16) {
      float* row = sc + lane * NK;
      float mx = -1e30f;
      for (int k = 0; k < NK; ++k) mx = fmaxf(mx, row[k]);
      float sum = 0.f;
      for (int k = 0; k < NK; ++k) { float e = __expf(row[k] - mx); row[k] = e; sum += e; }
      float inv = 1.f / sum;
      for (int k = 0; k < NK; ++k) row[k] *= inv;
    }
    __syncthreads();
    // ---- out(16x32) = P(16xNK) @ V ----
    v8f o2[2] = {};
    for (int k0 = 0; k0 < NK; k0 += 32) {
      v16h pa;
      {
        const float* row = sc + (lane & 15) * NK + k0 + ((lane >> 4) << 3);
#pragma unroll
        for (int t = 0; t < 8; ++t) { pa[t] = (h16)row[t]; pa[t + 8] = (h16)row[t + 16]; }
      }
#pragma unroll
      for (int t = 0; t < 2; ++t) {
        v16h vv = frag_ld(Vt + vt_off + (size_t)(t * 16) * NK + k0, NK);
        o2[t] = wmma16(pa, vv, o2[t]);
      }
    }
#pragma unroll
    for (int t = 0; t < 2; ++t)
#pragma unroll
      for (int r = 0; r < 8; ++r) {
        size_t oo = out_off + (size_t)(q0 + m_l + r) * out_rs + t * 16 + n_l;
        if (beta) out[oo] += o2[t][r]; else out[oo] = o2[t][r];
      }
    __syncthreads();
  }
}

// ---------------- host orchestration ----------------

extern "C" void kernel_launch(void* const* d_in, const int* in_sizes, int n_in,
                              void* d_out, int out_size, void* d_ws, size_t ws_size,
                              hipStream_t stream) {
  (void)in_sizes; (void)n_in; (void)out_size; (void)ws_size;
  const float* z      = (const float*)d_in[0];
  const float* zmask  = (const float*)d_in[1];
  const float* ppair  = (const float*)d_in[2];
  const float* mpair  = (const float*)d_in[3];
  const float* tri_ln_g  = (const float*)d_in[4];
  const float* tri_ln_b  = (const float*)d_in[5];
  const float* tri_lnh_g = (const float*)d_in[6];
  const float* tri_lnh_b = (const float*)d_in[7];
  const float* gate1_w = (const float*)d_in[8];   const float* gate1_b = (const float*)d_in[9];
  const float* gate2_w = (const float*)d_in[10];  const float* gate2_b = (const float*)d_in[11];
  const float* lin1_w  = (const float*)d_in[12];  const float* lin1_b  = (const float*)d_in[13];
  const float* lin2_w  = (const float*)d_in[14];  const float* lin2_b  = (const float*)d_in[15];
  const float* endg_w  = (const float*)d_in[16];  const float* endg_b  = (const float*)d_in[17];
  const float* asum_w  = (const float*)d_in[18];  const float* asum_b  = (const float*)d_in[19];
  const float* attn_ln_g = (const float*)d_in[20];
  const float* attn_ln_b = (const float*)d_in[21];
  const float* wq_w = (const float*)d_in[22];
  const float* wk_w = (const float*)d_in[23];
  const float* wv_w = (const float*)d_in[24];
  const float* wg_w = (const float*)d_in[25];  const float* wgb = (const float*)d_in[26];
  const float* wo_w = (const float*)d_in[27];  const float* wob = (const float*)d_in[28];
  const float* tr_ln_g = (const float*)d_in[29];
  const float* tr_ln_b = (const float*)d_in[30];
  const float* tr1_w = (const float*)d_in[31];  const float* tr1_b = (const float*)d_in[32];
  const float* tr2_w = (const float*)d_in[33];  const float* tr2_b = (const float*)d_in[34];

  char* ws = (char*)d_ws;
  size_t off = 0;
  auto alloc = [&](size_t bytes) -> char* {
    char* p = ws + off;
    off = (off + bytes + 255) & ~(size_t)255;
    return p;
  };
  const size_t S16_1 = (size_t)R1 * 128 * 2, S16_2 = (size_t)R2 * 128 * 2;
  const size_t S16_3 = (size_t)R3 * 128 * 2, S32_1 = (size_t)R1 * 128 * 4;

  h16* w16 = (h16*)alloc(245760 * 2);
  h16 *w_g1 = w16,          *w_g2 = w16 + 16384, *w_l1 = w16 + 32768,
      *w_l2 = w16 + 49152,  *w_eg = w16 + 65536, *w_as = w16 + 81920,
      *w_q  = w16 + 98304,  *w_k  = w16 + 114688, *w_v = w16 + 131072,
      *w_wg = w16 + 147456, *w_o  = w16 + 163840,
      *w_t1 = w16 + 180224, *w_t2 = w16 + 212992;
  h16* zn    = (h16*)alloc(S16_1);
  h16* ab1   = (h16*)alloc(S16_1);
  h16* ab2   = (h16*)alloc(S16_1);
  h16* gateb = (h16*)alloc(S16_1);
  h16* ppn   = (h16*)alloc(S16_2);
  h16* ppq   = (h16*)alloc(S16_2);
  h16* PPt   = (h16*)alloc(S16_2);
  h16* mpn   = (h16*)alloc(S16_3);
  h16* mpq   = (h16*)alloc(S16_3);
  h16* MPt   = (h16*)alloc(S16_3);
  h16* AB1t  = (h16*)alloc(S16_1);
  h16* AB2t  = (h16*)alloc(S16_1);
  float* bsum = (float*)alloc(S32_1);
  h16* hb    = (h16*)alloc(S16_1);
  float* z1  = (float*)alloc(S32_1);
  h16* zl    = (h16*)alloc(S16_1);
  h16* qb    = (h16*)alloc(S16_1);
  h16* kb    = (h16*)alloc(S16_1);
  h16* vb    = (h16*)alloc(S16_1);
  h16* gb    = (h16*)alloc(S16_1);
  h16* vT1   = (h16*)alloc(S16_1);
  h16* vT2   = (h16*)alloc(S16_1);
  float* att = (float*)alloc(S32_1);
  h16* zc    = (h16*)alloc(S16_1);
  float* z2  = (float*)alloc(S32_1);
  h16* zt    = (h16*)alloc(S16_1);
  h16* t1    = (h16*)alloc((size_t)R1 * 256 * 2);
  // total workspace ~455 MB

  dim3 B256(256), B128(128), B64(64);
  const int NE1 = R1 * 128;  // 6291456

  // weights -> f16
  struct { const float* s; h16* d; int n; } wj[13] = {
    {gate1_w, w_g1, 16384}, {gate2_w, w_g2, 16384}, {lin1_w, w_l1, 16384},
    {lin2_w, w_l2, 16384},  {endg_w, w_eg, 16384},  {asum_w, w_as, 16384},
    {wq_w, w_q, 16384},     {wk_w, w_k, 16384},     {wv_w, w_v, 16384},
    {wg_w, w_wg, 16384},    {wo_w, w_o, 16384},     {tr1_w, w_t1, 32768},
    {tr2_w, w_t2, 32768}};
  for (int i = 0; i < 13; ++i)
    k_f32_to_f16<<<dim3((wj[i].n + 255) / 256), B256, 0, stream>>>(wj[i].s, wj[i].d, wj[i].n);

  // ---- mol2pair ----
  k_ln<<<dim3(R1 / 8), B256, 0, stream>>>(z, tri_ln_g, tri_ln_b, zn, R1);
  k_ln<<<dim3(R2 / 8), B256, 0, stream>>>(ppair, tri_ln_g, tri_ln_b, ppn, R2);
  k_ln<<<dim3(R3 / 8), B256, 0, stream>>>(mpair, tri_ln_g, tri_ln_b, mpn, R3);

  k_dualgemm<<<dim3(R1 / 64, 4), B128, 0, stream>>>(zn, w_g1, gate1_b, w_l1, lin1_b, ab1, zmask);
  k_dualgemm<<<dim3(R1 / 64, 4), B128, 0, stream>>>(zn, w_g2, gate2_b, w_l2, lin2_b, ab2, zmask);
  k_dualgemm<<<dim3(R2 / 64, 4), B128, 0, stream>>>(ppn, w_g2, gate2_b, w_l2, lin2_b, ppq, nullptr);
  k_dualgemm<<<dim3(R3 / 64, 4), B128, 0, stream>>>(mpn, w_g1, gate1_b, w_l1, lin1_b, mpq, nullptr);
  k_gemm<1><<<dim3(R1 / 64, 2), B128, 0, stream>>>(zn, w_eg, endg_b, gateb, R1, 128, 128,
                                                   nullptr, nullptr, nullptr, 0.f);

  k_tr_cmajor<<<dim3(R2 * 128 / 256), B256, 0, stream>>>(ppq, PPt, R2);
  k_tr_cmajor<<<dim3(R1 * 128 / 256), B256, 0, stream>>>(ab2, AB2t, R1);
  k_tr_cmajor<<<dim3(R3 * 128 / 256), B256, 0, stream>>>(mpq, MPt, R3);
  k_perm_ab1<<<dim3(NE1 / 256), B256, 0, stream>>>(ab1, AB1t);

  k_tri<<<dim3(6, 2, 128), B128, 0, stream>>>(PPt, AB1t, bsum, 384, 0);   // pocket side
  k_tri<<<dim3(6, 2, 128), B128, 0, stream>>>(AB2t, MPt, bsum, 128, 1);   // mol side

  k_ln<<<dim3(R1 / 8), B256, 0, stream>>>(bsum, tri_lnh_g, tri_lnh_b, hb, R1);
  k_gemm<6><<<dim3(R1 / 64, 2), B128, 0, stream>>>(hb, w_as, asum_b, z1, R1, 128, 128,
                                                   z, gateb, zmask, 0.f);

  // ---- axis attention ----
  k_ln<<<dim3(R1 / 8), B256, 0, stream>>>(z1, attn_ln_g, attn_ln_b, zl, R1);
  k_gemm<3><<<dim3(R1 / 64, 2), B128, 0, stream>>>(zl, w_q, nullptr, qb, R1, 128, 128,
                                                   nullptr, nullptr, nullptr, 0.1767766953f);
  k_gemm<0><<<dim3(R1 / 64, 2), B128, 0, stream>>>(zl, w_k, nullptr, kb, R1, 128, 128,
                                                   nullptr, nullptr, nullptr, 0.f);
  k_gemm<0><<<dim3(R1 / 64, 2), B128, 0, stream>>>(zl, w_v, nullptr, vb, R1, 128, 128,
                                                   nullptr, nullptr, nullptr, 0.f);
  k_gemm<1><<<dim3(R1 / 64, 2), B128, 0, stream>>>(zl, w_wg, wgb, gb, R1, 128, 128,
                                                   nullptr, nullptr, nullptr, 0.f);

  k_perm_vT1<<<dim3(NE1 / 256), B256, 0, stream>>>(vb, vT1);
  k_perm_vT2<<<dim3(NE1 / 256), B256, 0, stream>>>(vb, vT2);

  k_attn<<<dim3(DI, 4), B64, 2 * 16 * 128 * 4, stream>>>(qb, kb, vT1, zmask, att, 1, 0);
  k_attn<<<dim3(DJ, 4), B64, 2 * 16 * 384 * 4, stream>>>(qb, kb, vT2, zmask, att, 2, 1);

  k_gmul<<<dim3(NE1 / 256), B256, 0, stream>>>(gb, att, zc, NE1);
  k_gemm<5><<<dim3(R1 / 64, 2), B128, 0, stream>>>(zc, w_o, wob, z2, R1, 128, 128,
                                                   z1, nullptr, zmask, 0.f);

  // ---- transition ----
  k_ln<<<dim3(R1 / 8), B256, 0, stream>>>(z2, tr_ln_g, tr_ln_b, zt, R1);
  k_gemm<2><<<dim3(R1 / 64, 4), B128, 0, stream>>>(zt, w_t1, tr1_b, t1, R1, 256, 128,
                                                   nullptr, nullptr, nullptr, 0.f);
  k_gemm<4><<<dim3(R1 / 64, 2), B128, 0, stream>>>(t1, w_t2, tr2_b, (float*)d_out,
                                                   R1, 128, 256, nullptr, nullptr, nullptr, 0.f);
}
